// S4D_7017976562395
// MI455X (gfx1250) — matrix operand order repaired and checked by reading, not compiled, not run on hardware
//
#include <hip/hip_runtime.h>
#include <hip/hip_bf16.h>
#include <math.h>

// Problem constants (match reference)
#define Hh   512
#define Nn   32
#define Bb   8
#define Ll   4096
#define L2X  8192            // 2L
#define ROWE 8208            // 2L + 16 elements, multiple of 8 (keeps 16B align per copy-row)

// Workspace layout (bytes)
#define WS_DTA   0u                       // dtA  complex f32 [H][N][2]      = 131072 B
#define WS_CTIL  131072u                  // Ctil complex f32 [2][H][N][2]   = 262144 B
#define WS_ZBUF  393216u                  // 64 bf16 zeros (padding lanes)
#define WS_UBF   524288u                  // bf16 u  [B][H][L]               = 33554432 B
#define WS_REV   34078720u                // bf16 Rev[H][8][ROWE]            = 67239936 B
#define WS_NEED  101318656u

typedef __attribute__((ext_vector_type(16))) __bf16 v16bf;
typedef __attribute__((ext_vector_type(8)))  __bf16 v8bf;
typedef __attribute__((ext_vector_type(8)))  float  v8f;

// ---------------------------------------------------------------------------
// Kernel 0: ZOH-like discretization.  dtA = dt*A,  Ctil = C*B*(exp(dtA)-1)/A
// ---------------------------------------------------------------------------
__global__ __launch_bounds__(256)
void s4d_coeffs(const float* __restrict__ log_dt, const float* __restrict__ log_neg_A_re,
                const float* __restrict__ A_im,   const float* __restrict__ B_ri,
                const float* __restrict__ C_ri,   float* __restrict__ dtA,
                float* __restrict__ Ctil) {
    int idx = blockIdx.x * blockDim.x + threadIdx.x;      // h*N + n
    if (idx >= Hh * Nn) return;
    int h = idx / Nn;
    float dt  = __expf(log_dt[h]);
    float Are = -__expf(log_neg_A_re[idx]);
    float Aim = A_im[idx];
    float dr = dt * Are, di = dt * Aim;
    dtA[2 * idx]     = dr;
    dtA[2 * idx + 1] = di;
    float Br = B_ri[2 * idx], Bi = B_ri[2 * idx + 1];
    // E = exp(dtA) - 1
    float ex = __expf(dr);
    float sn, cs; __sincosf(di, &sn, &cs);
    float Er = ex * cs - 1.0f;
    float Ei = ex * sn;
    // M = E / A  (complex divide)
    float inv = 1.0f / (Are * Are + Aim * Aim);
    float Mr = (Er * Are + Ei * Aim) * inv;
    float Mi = (Ei * Are - Er * Aim) * inv;
    #pragma unroll
    for (int c = 0; c < 2; ++c) {
        const float* Cp = C_ri + (size_t)(c * Hh * Nn + idx) * 2;
        float Cr = Cp[0], Ci = Cp[1];
        float CBr = Cr * Br - Ci * Bi;
        float CBi = Cr * Bi + Ci * Br;
        float* Tp = Ctil + (size_t)(c * Hh * Nn + idx) * 2;
        Tp[0] = CBr * Mr - CBi * Mi;
        Tp[1] = CBr * Mi + CBi * Mr;
    }
}

// ---------------------------------------------------------------------------
// Kernel 1: evaluate combined kernel Kc[d] = 2*Re(sum_n Ctil[c]*exp(dtA*lp))
// (d>=0 -> causal k0[d]; d<0 -> anticausal k1[-d-1]) and scatter it, reversed,
// into 8 element-shifted bf16 copies so the WMMA kernel gets 16B-aligned b128
// loads for every Toeplitz row.
//   RevAll[h][c8][ (L-1-d) + c8 ] = Kc[h][d]
// ---------------------------------------------------------------------------
__global__ __launch_bounds__(256)
void s4d_build_rev(const float* __restrict__ dtA, const float* __restrict__ Ctil,
                   unsigned short* __restrict__ rev_us) {
    int idx = blockIdx.x * blockDim.x + threadIdx.x;      // h*2L + j
    if (idx >= Hh * L2X) return;
    int h = idx >> 13;                 // / 8192
    int j = idx & (L2X - 1);
    int d = j - Ll;                    // [-L, L-1]
    int c = (d >= 0) ? 0 : 1;
    float lp = (float)((d >= 0) ? d : (-d - 1));
    const float* dA = dtA  + (size_t)h * Nn * 2;
    const float* Ct = Ctil + ((size_t)c * Hh + h) * Nn * 2;
    float acc = 0.0f;
    #pragma unroll 4
    for (int n = 0; n < Nn; ++n) {
        float dr = dA[2 * n], di = dA[2 * n + 1];
        float er = __expf(dr * lp);
        float sn, cs; __sincosf(di * lp, &sn, &cs);
        acc += er * (Ct[2 * n] * cs - Ct[2 * n + 1] * sn);
    }
    acc *= 2.0f;
    __bf16 v = (__bf16)acc;
    __bf16* row = reinterpret_cast<__bf16*>(rev_us) + (size_t)h * 8 * ROWE;
    int pos0 = (Ll - 1) - d;           // reversed position, [0, 2L)
    #pragma unroll
    for (int c8 = 0; c8 < 8; ++c8)
        row[(size_t)c8 * ROWE + pos0 + c8] = v;
}

// ---------------------------------------------------------------------------
// Kernel 2: u (f32) -> bf16, plus a small zero block for padding lanes (b>=8).
// ---------------------------------------------------------------------------
__global__ __launch_bounds__(256)
void s4d_pack_u(const float* __restrict__ u, unsigned short* __restrict__ ubf_us,
                unsigned short* __restrict__ zbuf_us) {
    int idx = blockIdx.x * blockDim.x + threadIdx.x;
    if (blockIdx.x == 0 && idx < 64)
        reinterpret_cast<__bf16*>(zbuf_us)[idx] = (__bf16)0.0f;
    if (idx >= Bb * Hh * Ll) return;
    reinterpret_cast<__bf16*>(ubf_us)[idx] = (__bf16)u[idx];
}

// ---------------------------------------------------------------------------
// Kernel 3: Toeplitz GEMM via v_wmma_f32_16x16x32_bf16.
//   A (16x32, M=l rows)  = Kc[l - t] tiles, streamed from the shifted copies
//   B (32x16, N=b cols)  = u bf16 (cols b>=8 read a zero block)
//   D (16x16 f32)        = y^T tile; epilogue adds D[h]*u.
// One wave owns two adjacent 16-row l-tiles (shares the B operand): inner loop
// is 6x global_load_b128 + 2x WMMA per 32 t-steps. Both operand streams are
// L2-resident (32MB + 64MB < 192MB).
// ---------------------------------------------------------------------------
__global__ __launch_bounds__(256)
void s4d_wmma_conv(const float* __restrict__ u, const float* __restrict__ Dv,
                   const unsigned short* __restrict__ ubf_us,
                   const unsigned short* __restrict__ rev_us,
                   const unsigned short* __restrict__ zbuf_us,
                   float* __restrict__ y) {
    const __bf16* ubf  = reinterpret_cast<const __bf16*>(ubf_us);
    const __bf16* Rev  = reinterpret_cast<const __bf16*>(rev_us);
    const __bf16* zbuf = reinterpret_cast<const __bf16*>(zbuf_us);

    int tid  = threadIdx.x;
    int wave = blockIdx.x * (blockDim.x >> 5) + (tid >> 5);
    int lane = tid & 31;
    int h  = wave >> 7;                 // 128 double-tiles per h
    int lg = wave & 127;
    int l0 = lg << 5;                   // rows l0..l0+31 (two 16-row tiles)
    int lr = lane & 15;
    int hi = lane >> 4;                 // K-half select

    // A operand: Toeplitz rows. copy c = (l+1)&7 makes (L-1-l)+c = 0 mod 8.
    int l = l0 + lr;
    int c = (l + 1) & 7;
    const __bf16* pA0 = Rev + ((size_t)(h * 8 + c) * ROWE)
                            + (size_t)((Ll - 1) - l + c + 8 * hi);
    const __bf16* pA1 = pA0 - 16;       // tile1: row l+16 -> base shifts by -16

    // B operand: u columns (b = lr). Padding lanes stream a 64-elem zero block.
    bool valid = (lr < Bb);
    const __bf16* pB = valid ? (ubf + ((size_t)(lr * Hh + h) * Ll) + 8 * hi)
                             : zbuf;
    size_t sB = valid ? 32 : 0;

    v8f acc0 = {};
    v8f acc1 = {};
    #pragma unroll 4
    for (int t0 = 0; t0 < Ll; t0 += 32) {
        v16bf a0, a1, b;
        *(v8bf*)&a0       = *(const v8bf*)(pA0);
        *((v8bf*)&a0 + 1) = *(const v8bf*)(pA0 + 16);
        *(v8bf*)&a1       = *(const v8bf*)(pA1);
        *((v8bf*)&a1 + 1) = *(const v8bf*)(pA1 + 16);
        *(v8bf*)&b        = *(const v8bf*)(pB);
        *((v8bf*)&b + 1)  = *(const v8bf*)(pB + 16);
        acc0 = __builtin_amdgcn_wmma_f32_16x16x32_bf16(false, a0, false, b,
                                                       (short)0, acc0, false, false);
        acc1 = __builtin_amdgcn_wmma_f32_16x16x32_bf16(false, a1, false, b,
                                                       (short)0, acc1, false, false);
        pA0 += 32; pA1 += 32; pB += sB;
    }

    // Epilogue: D tile -> lane holds column b=lr, VGPR r -> row l0 + 8*hi + r.
    if (valid) {
        int lb0 = l0 + 8 * hi;
        const float* pu = u + ((size_t)(lr * Hh + h) * Ll);
        float*       py = y + ((size_t)(lr * Hh + h) * Ll);
        float dh = Dv[h];
        #pragma unroll
        for (int r = 0; r < 8; ++r) {
            int li = lb0 + r;
            py[li]      = acc0[r] + dh * pu[li];
            py[li + 16] = acc1[r] + dh * pu[li + 16];
        }
    }
}

// ---------------------------------------------------------------------------
extern "C" void kernel_launch(void* const* d_in, const int* in_sizes, int n_in,
                              void* d_out, int out_size, void* d_ws, size_t ws_size,
                              hipStream_t stream) {
    (void)in_sizes; (void)n_in; (void)out_size;
    if (ws_size < (size_t)WS_NEED) return;   // workspace too small: nothing safe to do

    const float* u            = (const float*)d_in[0];
    const float* log_dt       = (const float*)d_in[1];
    const float* log_neg_A_re = (const float*)d_in[2];
    const float* A_im         = (const float*)d_in[3];
    const float* B_ri         = (const float*)d_in[4];
    const float* C_ri         = (const float*)d_in[5];
    const float* Dv           = (const float*)d_in[6];
    float* y = (float*)d_out;

    char* ws = (char*)d_ws;
    float*          dtA  = (float*)(ws + WS_DTA);
    float*          Ctil = (float*)(ws + WS_CTIL);
    unsigned short* zbuf = (unsigned short*)(ws + WS_ZBUF);
    unsigned short* ubf  = (unsigned short*)(ws + WS_UBF);
    unsigned short* rev  = (unsigned short*)(ws + WS_REV);

    // 0) discretize (16K threads)
    s4d_coeffs<<<(Hh * Nn + 255) / 256, 256, 0, stream>>>(
        log_dt, log_neg_A_re, A_im, B_ri, C_ri, dtA, Ctil);

    // 1) kernel table -> 8 shifted reversed bf16 copies (4.2M threads)
    s4d_build_rev<<<(Hh * L2X) / 256, 256, 0, stream>>>(dtA, Ctil, rev);

    // 2) u -> bf16 (16.7M threads)
    s4d_pack_u<<<(Bb * Hh * Ll) / 256, 256, 0, stream>>>(u, ubf, zbuf);

    // 3) WMMA Toeplitz GEMM: 65536 waves, 8 waves/block
    s4d_wmma_conv<<<(Hh * 128) / 8, 256, 0, stream>>>(u, Dv, ubf, rev, zbuf, y);
}
